// MultiScaleRetention_16441134809482
// MI455X (gfx1250) — matrix operand order repaired and checked by reading, not compiled, not run on hardware
//
#include <hip/hip_runtime.h>
#include <hip/hip_bf16.h>
#include <stdint.h>

// ---------------- problem constants (fixed by reference) ----------------
#define BB   4
#define LL   4096
#define DD   2048
#define HH   16
#define DHH  128
#define MTOT (BB * LL)          // 16384 rows

// ---------------- WMMA vector types (gfx1250, wave32) -------------------
typedef __attribute__((ext_vector_type(16))) __bf16 v16bf;
typedef __attribute__((ext_vector_type(8)))  float  v8f;

// ---------------- helpers ----------------------------------------------
__device__ __forceinline__ uint16_t f32_to_bf16_rne(float f) {
    union { float f; uint32_t u; } v; v.f = f;
    uint32_t u = v.u;
    if ((u & 0x7fffffffu) > 0x7f800000u) return (uint16_t)((u >> 16) | 0x40); // qNaN
    u += 0x7fffu + ((u >> 16) & 1u);
    return (uint16_t)(u >> 16);
}

__global__ void convert_f32_bf16(const float* __restrict__ in,
                                 uint16_t* __restrict__ out, int n) {
    int i = blockIdx.x * blockDim.x + threadIdx.x;
    int stride = gridDim.x * blockDim.x;
    for (; i < n; i += stride) out[i] = f32_to_bf16_rne(in[i]);
}

// ---------------- bf16 WMMA GEMM:  C[M,N] = A[M,K] * W[N,K]^T -----------
// 256 threads = 8 waves; block tile 128x128, K-step 32; double-buffered LDS
// filled with CDNA5 async global->LDS copies (ASYNCcnt), WMMA on bf16.
#define BM 128
#define BN 128
#define BK 32
#define LDSS 40   // padded row stride in bf16 elems (80B) -> bank-conflict relief

__global__ __launch_bounds__(256)
void gemm_bf16_wmma(const uint16_t* __restrict__ A,   // M x K row-major bf16
                    const uint16_t* __restrict__ W,   // N x K row-major bf16
                    float* __restrict__ C,            // M x N row-major f32
                    int M, int N, int K)
{
    __shared__ __align__(16) uint16_t sA[2][BM * LDSS];
    __shared__ __align__(16) uint16_t sB[2][BN * LDSS];

    const int tid   = threadIdx.x;
    const int lane  = tid & 31;
    const int wid   = tid >> 5;       // 8 waves
    const int wm    = wid & 3;        // 4 waves along M: 32 rows each
    const int wn    = wid >> 2;       // 2 waves along N: 64 cols each
    const int lrow  = lane & 15;
    const int lhalf = lane >> 4;      // 0 / 1

    const int bm = blockIdx.y * BM;
    const int bn = blockIdx.x * BN;

    // --- staging geometry: each thread moves 2 x 16B chunks for A and for B.
    // chunk0: row = tid/4,     col = (tid%4)*8 ; chunk1: row += 64, same col.
    const int srow = tid >> 2;
    const int skc  = (tid & 3) << 3;
    const uint16_t* gA = &A[(size_t)(bm + srow) * K + skc];
    const uint16_t* gW = &W[(size_t)(bn + srow) * K + skc];
    const size_t rowHop = (size_t)64 * K;          // +64 rows in global
    const uint32_t ldsHop = 64u * LDSS * 2u;       // +64 rows in LDS (bytes)

    v8f acc[2][4];
    #pragma unroll
    for (int mi = 0; mi < 2; ++mi)
        #pragma unroll
        for (int ni = 0; ni < 4; ++ni)
            #pragma unroll
            for (int e = 0; e < 8; ++e) acc[mi][ni][e] = 0.0f;

    const int khalfA = lhalf << 4;    // 0 or 16  (contiguous k order, same as B)
    const int khalfB = lhalf << 4;    // 0 or 16  (contiguous k order, same as A)

    // async global->LDS stage of one 128x32 tile pair into buffer `buf`
    auto stage = [&](int buf, int k0) {
        uint32_t laA = (uint32_t)(uintptr_t)&sA[buf][srow * LDSS + skc];
        uint32_t laB = (uint32_t)(uintptr_t)&sB[buf][srow * LDSS + skc];
        const uint16_t* pA = gA + k0;
        const uint16_t* pW = gW + k0;
        asm volatile("global_load_async_to_lds_b128 %0, %1, off"
                     :: "v"(laA), "v"(pA) : "memory");
        asm volatile("global_load_async_to_lds_b128 %0, %1, off"
                     :: "v"(laB), "v"(pW) : "memory");
        asm volatile("global_load_async_to_lds_b128 %0, %1, off"
                     :: "v"(laA + ldsHop), "v"(pA + rowHop) : "memory");
        asm volatile("global_load_async_to_lds_b128 %0, %1, off"
                     :: "v"(laB + ldsHop), "v"(pW + rowHop) : "memory");
    };

    stage(0, 0);
    asm volatile("s_wait_asynccnt 0x0" ::: "memory");
    __syncthreads();

    int buf = 0;
    for (int k0 = 0; k0 < K; k0 += BK) {
        // prefetch tile k0+1 asynchronously into the other buffer
        if (k0 + BK < K) stage(buf ^ 1, k0 + BK);
        // warm L2 two tiles ahead (global_prefetch_b8)
        if (k0 + 2 * BK < K) {
            __builtin_prefetch(gA + k0 + 2 * BK, 0, 2);
            __builtin_prefetch(gW + k0 + 2 * BK, 0, 2);
        }

        // A fragments: lane holds row (lrow), K = khalfA..khalfA+15 (contiguous)
        v16bf afrag[2], bfrag[4];
        #pragma unroll
        for (int mi = 0; mi < 2; ++mi) {
            const uint16_t* p = &sA[buf][(wm * 32 + mi * 16 + lrow) * LDSS];
            uint4* ap = reinterpret_cast<uint4*>(&afrag[mi]);
            ap[0] = *(const uint4*)(&p[khalfA]);
            ap[1] = *(const uint4*)(&p[khalfA + 8]);
        }
        // B fragments: lane n holds W[n, khalfB .. khalfB+15] (contiguous)
        #pragma unroll
        for (int ni = 0; ni < 4; ++ni) {
            const uint16_t* p = &sB[buf][(wn * 64 + ni * 16 + lrow) * LDSS + khalfB];
            uint4* bp = reinterpret_cast<uint4*>(&bfrag[ni]);
            bp[0] = *(const uint4*)(&p[0]);
            bp[1] = *(const uint4*)(&p[8]);
        }

        #pragma unroll
        for (int mi = 0; mi < 2; ++mi)
            #pragma unroll
            for (int ni = 0; ni < 4; ++ni)
                acc[mi][ni] = __builtin_amdgcn_wmma_f32_16x16x32_bf16(
                    /*neg_a=*/false, afrag[mi],
                    /*neg_b=*/false, bfrag[ni],
                    /*c_mod=*/(short)0, acc[mi][ni],
                    /*reuse_a=*/false, /*reuse_b=*/false);

        // my async copies into buf^1 done; barrier publishes them to all waves
        // and guarantees everyone finished reading `buf` (wmma forced dscnt).
        asm volatile("s_wait_asynccnt 0x0" ::: "memory");
        __syncthreads();
        buf ^= 1;
    }

    // epilogue: C/D 16x16 f32 layout: VGPR r -> M = r + 8*lhalf, N = lrow
    #pragma unroll
    for (int mi = 0; mi < 2; ++mi)
        #pragma unroll
        for (int ni = 0; ni < 4; ++ni) {
            const int col   = bn + wn * 64 + ni * 16 + lrow;
            const int rbase = bm + wm * 32 + mi * 16 + lhalf * 8;
            #pragma unroll
            for (int r = 0; r < 8; ++r)
                C[(size_t)(rbase + r) * N + col] = acc[mi][ni][r];
        }
}

// ---------------- retention scan (in-place on v): s_t = lam*s_{t-1}+v_t --
__global__ void retention_scan(float* __restrict__ v,
                               const float* __restrict__ beta) {
    int idx = blockIdx.x * blockDim.x + threadIdx.x;  // channel = (b, h*Dh+dh)
    if (idx >= BB * DD) return;
    int b  = idx / DD;
    int hd = idx - b * DD;
    int h  = hd / DHH;
    float lam = 1.0f / (1.0f + __expf(-beta[h]));
    float s = 0.0f;
    float* p = v + (size_t)b * LL * DD + hd;          // stride over t is DD floats
    #pragma unroll 4
    for (int t = 0; t < LL; ++t) {
        s = fmaf(lam, s, p[(size_t)t * DD]);
        p[(size_t)t * DD] = s;
    }
}

// ---------------- LayerNorm + SiLU gate, emits bf16 y --------------------
__global__ __launch_bounds__(256)
void ln_gate_kernel(const float* __restrict__ q, const float* __restrict__ st,
                    const float* __restrict__ g, const float* __restrict__ lnw,
                    const float* __restrict__ lnb, uint16_t* __restrict__ yb)
{
    const int row = blockIdx.x;                       // 0..MTOT-1
    const int tid = threadIdx.x;                      // 256 threads, 8 elems each
    const float* qr = q  + (size_t)row * DD;
    const float* sr = st + (size_t)row * DD;
    const float* gr = g  + (size_t)row * DD;

    float p[8];
    float sum = 0.0f, sq = 0.0f;
    #pragma unroll
    for (int i = 0; i < 8; ++i) {
        int d = tid + i * 256;
        float t = qr[d] * sr[d];
        p[i] = t; sum += t; sq += t * t;
    }
    // wave32 reduction
    #pragma unroll
    for (int off = 16; off > 0; off >>= 1) {
        sum += __shfl_xor(sum, off, 32);
        sq  += __shfl_xor(sq,  off, 32);
    }
    __shared__ float red[2][8];
    const int lane = tid & 31, wid = tid >> 5;
    if (lane == 0) { red[0][wid] = sum; red[1][wid] = sq; }
    __syncthreads();
    float tsum = 0.0f, tsq = 0.0f;
    #pragma unroll
    for (int w = 0; w < 8; ++w) { tsum += red[0][w]; tsq += red[1][w]; }

    const float inv  = 1.0f / (float)DD;
    const float mean = tsum * inv;
    const float var  = tsq * inv - mean * mean;
    const float rstd = rsqrtf(var + 1e-5f);

    #pragma unroll
    for (int i = 0; i < 8; ++i) {
        int d = tid + i * 256;
        float gv   = gr[d];
        float gate = gv / (1.0f + __expf(-gv));       // silu
        float yv   = (p[i] - mean) * rstd * lnw[d] + lnb[d];
        yb[(size_t)row * DD + d] = f32_to_bf16_rne(yv * gate);
    }
}

// ---------------- orchestration -----------------------------------------
extern "C" void kernel_launch(void* const* d_in, const int* in_sizes, int n_in,
                              void* d_out, int out_size, void* d_ws, size_t ws_size,
                              hipStream_t stream) {
    (void)in_sizes; (void)n_in; (void)out_size; (void)ws_size;
    const float* x    = (const float*)d_in[0];
    const float* Wq   = (const float*)d_in[1];
    const float* Wv   = (const float*)d_in[2];
    const float* Wg   = (const float*)d_in[3];
    const float* Wo   = (const float*)d_in[4];
    const float* beta = (const float*)d_in[5];
    const float* lnw  = (const float*)d_in[6];
    const float* lnb  = (const float*)d_in[7];
    float* out = (float*)d_out;

    const size_t nX = (size_t)MTOT * DD;   // 33,554,432
    const size_t nW = (size_t)DD * DD;     //  4,194,304

    uint8_t* ws = (uint8_t*)d_ws;
    uint16_t* xb  = (uint16_t*)ws; ws += nX * 2;   // x   bf16
    uint16_t* wqb = (uint16_t*)ws; ws += nW * 2;   // Wq  bf16
    uint16_t* wvb = (uint16_t*)ws; ws += nW * 2;   // Wv  bf16
    uint16_t* wgb = (uint16_t*)ws; ws += nW * 2;   // Wg  bf16
    uint16_t* wob = (uint16_t*)ws; ws += nW * 2;   // Wo  bf16
    float*    qf  = (float*)ws;    ws += nX * 4;   // q   f32
    float*    vf  = (float*)ws;    ws += nX * 4;   // v -> state (in place)
    float*    gf  = (float*)ws;    ws += nX * 4;   // g   f32
    uint16_t* yb  = (uint16_t*)ws; ws += nX * 2;   // y   bf16

    // 1) precision conversion
    convert_f32_bf16<<<4096, 256, 0, stream>>>(x,  xb,  (int)nX);
    convert_f32_bf16<<<2048, 256, 0, stream>>>(Wq, wqb, (int)nW);
    convert_f32_bf16<<<2048, 256, 0, stream>>>(Wv, wvb, (int)nW);
    convert_f32_bf16<<<2048, 256, 0, stream>>>(Wg, wgb, (int)nW);
    convert_f32_bf16<<<2048, 256, 0, stream>>>(Wo, wob, (int)nW);

    // 2) q/v/g projections (WMMA GEMMs)
    dim3 ggrid(DD / BN, MTOT / BM);                // (16, 128)
    gemm_bf16_wmma<<<ggrid, 256, 0, stream>>>(xb, wqb, qf, MTOT, DD, DD);
    gemm_bf16_wmma<<<ggrid, 256, 0, stream>>>(xb, wvb, vf, MTOT, DD, DD);
    gemm_bf16_wmma<<<ggrid, 256, 0, stream>>>(xb, wgb, gf, MTOT, DD, DD);

    // 3) per-head decay scan (in place on vf)
    retention_scan<<<(BB * DD + 255) / 256, 256, 0, stream>>>(vf, beta);

    // 4) y = LN(q*state) * silu(g)  ->  bf16
    ln_gate_kernel<<<MTOT, 256, 0, stream>>>(qf, vf, gf, lnw, lnb, yb);

    // 5) output projection
    gemm_bf16_wmma<<<ggrid, 256, 0, stream>>>(yb, wob, out, MTOT, DD, DD);
}
